// TerrainEncoder_51135880626706
// MI455X (gfx1250) — matrix operand order repaired and checked by reading, not compiled, not run on hardware
//
#include <hip/hip_runtime.h>

typedef __attribute__((ext_vector_type(16))) _Float16 v16h;
typedef __attribute__((ext_vector_type(8)))  _Float16 v8h;
typedef __attribute__((ext_vector_type(8)))  float    v8f;

namespace {
constexpr int   kB   = 4;
constexpr int   kN   = 200000;
constexpr int   kC   = 8;
constexpr float kPS  = 0.5f;
constexpr float kXMIN = -60.f;
constexpr float kYMIN = -60.f;
constexpr int   kW   = 240;
constexpr int   kH   = 240;
constexpr int   kP   = kW * kH;       // 57600 pillars
constexpr int   kD   = 64;            // PFN out channels
constexpr int   kFD  = 256;
constexpr int   kAUG = 13;
constexpr float kEPS = 1e-5f;
}

// ---------------------------------------------------------------------------
// Stage 1a: scatter-add per-pillar xyz sums and counts
// ---------------------------------------------------------------------------
__global__ void scatter_mean_kernel(const float* __restrict__ pts,
                                    float* __restrict__ psum,
                                    float* __restrict__ pcnt) {
    int i = blockIdx.x * blockDim.x + threadIdx.x;
    if (i >= kB * kN) return;
    const float* pt = pts + (size_t)i * kC;
    float x = pt[0], y = pt[1];
    int px = (int)((x - kXMIN) / kPS);  // trunc toward zero == jnp.trunc
    int py = (int)((y - kYMIN) / kPS);
    if (px < 0 || px >= kW || py < 0 || py >= kH) return;
    int b = i / kN;
    int seg = b * kP + py * kW + px;
    atomicAdd(&psum[(size_t)seg * 3 + 0], pt[0]);
    atomicAdd(&psum[(size_t)seg * 3 + 1], pt[1]);
    atomicAdd(&psum[(size_t)seg * 3 + 2], pt[2]);
    atomicAdd(&pcnt[seg], 1.0f);
}

// ---------------------------------------------------------------------------
// Stage 1b: PFN (13 -> 64 linear + BN + ReLU) + segment max into pillar grid.
// All maxed values are >= 0 so uint-bit atomicMax against a 0-initialized
// buffer is order-correct.
// ---------------------------------------------------------------------------
__global__ __launch_bounds__(256)
void pfn_max_kernel(const float* __restrict__ pts,
                    const float* __restrict__ psum,
                    const float* __restrict__ pcnt,
                    const float* __restrict__ w,     // (64, 13)
                    const float* __restrict__ pb,    // (64,)
                    const float* __restrict__ bg,
                    const float* __restrict__ bbias,
                    const float* __restrict__ bm,
                    const float* __restrict__ bv,
                    float* __restrict__ pmax) {      // (B, P, 64), init 0
    __shared__ float sw[kD * kAUG];
    __shared__ float sscale[kD];
    __shared__ float sshift[kD];
    int tid = threadIdx.x;
    for (int i = tid; i < kD * kAUG; i += 256) sw[i] = w[i];
    if (tid < kD) {
        float s = bg[tid] * rsqrtf(bv[tid] + kEPS);
        sscale[tid] = s;
        sshift[tid] = (pb[tid] - bm[tid]) * s + bbias[tid];
    }
    __syncthreads();

    int i = blockIdx.x * blockDim.x + tid;
    if (i >= kB * kN) return;
    const float* pt = pts + (size_t)i * kC;
    float x = pt[0], y = pt[1];
    int px = (int)((x - kXMIN) / kPS);
    int py = (int)((y - kYMIN) / kPS);
    if (px < 0 || px >= kW || py < 0 || py >= kH) return;  // invalid -> zero row
    int b = i / kN;
    int seg = b * kP + py * kW + px;

    float cnt = fmaxf(pcnt[seg], 1.0f);
    float mx = psum[(size_t)seg * 3 + 0] / cnt;
    float my = psum[(size_t)seg * 3 + 1] / cnt;
    float mz = psum[(size_t)seg * 3 + 2] / cnt;
    float cx = kXMIN + ((float)px + 0.5f) * kPS;
    float cy = kYMIN + ((float)py + 0.5f) * kPS;

    float aug[kAUG];
#pragma unroll
    for (int k = 0; k < kC; ++k) aug[k] = pt[k];
    aug[8]  = pt[0] - mx;
    aug[9]  = pt[1] - my;
    aug[10] = pt[2] - mz;
    aug[11] = x - cx;
    aug[12] = y - cy;

    unsigned int* dst = (unsigned int*)(pmax + (size_t)seg * kD);
#pragma unroll 4
    for (int d = 0; d < kD; ++d) {
        float dot = 0.f;
#pragma unroll
        for (int k = 0; k < kAUG; ++k) dot += sw[d * kAUG + k] * aug[k];
        float f = fmaxf(dot * sscale[d] + sshift[d], 0.f);
        atomicMax(&dst[d], __float_as_uint(f));
    }
}

// ---------------------------------------------------------------------------
// Stage 2a: pillar grid (B, P, 64) f32 is ALREADY NHWC (B, H, W, 64):
// just cast f32 -> f16, fully coalesced.
// ---------------------------------------------------------------------------
__global__ void bev_cast_kernel(const float* __restrict__ pmax,
                                _Float16* __restrict__ bev) {
    size_t idx = (size_t)blockIdx.x * blockDim.x + threadIdx.x;
    if (idx >= (size_t)kB * kP * kD) return;
    bev[idx] = (_Float16)pmax[idx];
}

// ---------------------------------------------------------------------------
// Weight prep: OIHW f32 -> [tap][cout][cin] f16 (contiguous in cin for
// A-fragment vector loads)
// ---------------------------------------------------------------------------
__global__ void wperm_kernel(const float* __restrict__ src,
                             _Float16* __restrict__ dst,
                             int CIN, int COUT) {
    int idx = blockIdx.x * blockDim.x + threadIdx.x;
    int total = COUT * CIN * 9;
    if (idx >= total) return;
    int co = idx / (CIN * 9);
    int r  = idx - co * CIN * 9;
    int ci = r / 9;
    int t  = r - ci * 9;
    dst[((size_t)t * COUT + co) * CIN + ci] = (_Float16)src[idx];
}

// ---------------------------------------------------------------------------
// Stage 2b: 3x3 conv + bias + BN + ReLU via V_WMMA_F32_16X16X32_F16, NHWC.
// One wave per block: 16 couts x 64 pixels (4 accumulators, A reused 4x).
// Tap-decomposed implicit GEMM: 9 taps x (CIN/32) K-chunks; K loop fully
// unrolled; per-tap byte pointers precomputed so the constant kc offset
// folds into the global_load immediate (no per-fragment address VALU);
// load-phase / WMMA-phase split widens load->use and WMMA->overwrite
// distances (kills WAR-hazard v_nops).
// Fragment layouts used here (wave32; contiguous K, same order in A and B):
//   A 16x32 f16 : M = lane&15, K = e + 16*(lane>=16)
//   B 32x16 f16 : N = lane&15, K = e + 16*(lane>=16)
//   C 16x16 f32 : M = r + 8*(lane>=16), N = lane&15
// ---------------------------------------------------------------------------
template <int CIN, int COUT>
__global__ __launch_bounds__(32)
void conv3x3_wmma(const _Float16* __restrict__ in,     // (B, Hin, Win, CIN)
                  const _Float16* __restrict__ wp,     // (9, COUT, CIN)
                  const float* __restrict__ cb,
                  const float* __restrict__ g,
                  const float* __restrict__ bb,
                  const float* __restrict__ bm,
                  const float* __restrict__ bv,
                  _Float16* __restrict__ out,          // (B, Hout, Wout, COUT)
                  int Hin, int Win, int Hout, int Wout, int stride) {
    const int lane  = threadIdx.x;
    const int lo    = lane & 15;
    const int hi    = lane >> 4;         // 0/1: lane half
    const int HWo   = Hout * Wout;
    const int pBase = blockIdx.x * 64;
    const int mBase = blockIdx.y * 16;
    const int b     = blockIdx.z;

    // Per-lane output pixel coords (4 sub-tiles of 16 pixels each)
    int  oy0[4], ox0[4];
    bool pin[4];
#pragma unroll
    for (int t = 0; t < 4; ++t) {
        int p = pBase + t * 16 + lo;
        pin[t] = (p < HWo);
        int pp = pin[t] ? p : 0;
        oy0[t] = (pp / Wout) * stride - 1;
        ox0[t] = (pp - (pp / Wout) * Wout) * stride - 1;
    }

    const _Float16* inb = in + (size_t)b * Hin * Win * CIN + 16 * hi;
    const v16h bzero = {};
    v8f acc[4] = {};

    for (int tap = 0; tap < 9; ++tap) {
        const int ky = tap / 3, kx = tap % 3;
        // Per-sub-tile: clamped byte pointer + validity (hoisted out of K loop)
        const _Float16* bp[4];
        bool bok[4];
#pragma unroll
        for (int t = 0; t < 4; ++t) {
            int iy = oy0[t] + ky;
            int ix = ox0[t] + kx;
            bok[t] = pin[t] & (iy >= 0) & (iy < Hin) & (ix >= 0) & (ix < Win);
            int iyc = iy < 0 ? 0 : (iy >= Hin ? Hin - 1 : iy);
            int ixc = ix < 0 ? 0 : (ix >= Win ? Win - 1 : ix);
            bp[t] = inb + ((size_t)iyc * Win + ixc) * CIN;
        }
        const _Float16* wt = wp + (size_t)tap * COUT * CIN + (size_t)(mBase + lo) * CIN + 16 * hi;
#pragma unroll
        for (int kc = 0; kc < CIN; kc += 32) {
            // A fragment: two contiguous 8-half chunks (16B vector loads,
            // kc folds into the load immediate offset)
            v8h a0 = *(const v8h*)(wt + kc);
            v8h a1 = *(const v8h*)(wt + kc + 8);
            v16h a = __builtin_shufflevector(a0, a1, 0, 1, 2, 3, 4, 5, 6, 7,
                                             8, 9, 10, 11, 12, 13, 14, 15);
            // Load phase: all 4 B fragments into distinct register quads
            v16h bf[4];
#pragma unroll
            for (int t = 0; t < 4; ++t) {
                v16h bv16 = *(const v16h*)(bp[t] + kc);
                bf[t] = bok[t] ? bv16 : bzero;
            }
            // WMMA phase
#pragma unroll
            for (int t = 0; t < 4; ++t)
                acc[t] = __builtin_amdgcn_wmma_f32_16x16x32_f16(
                    false, a, false, bf[t], (short)0, acc[t], false, false);
        }
    }

    // Epilogue: bias + BN + ReLU, 8 contiguous halves per lane -> b128 store
    float sA[8], sB[8];
#pragma unroll
    for (int r = 0; r < 8; ++r) {
        int co = mBase + 8 * hi + r;
        float s = g[co] * rsqrtf(bv[co] + kEPS);
        sA[r] = s;
        sB[r] = (cb[co] - bm[co]) * s + bb[co];
    }
    _Float16* outb = out + (size_t)b * HWo * COUT + mBase + 8 * hi;
#pragma unroll
    for (int t = 0; t < 4; ++t) {
        int p = pBase + t * 16 + lo;
        if (p < HWo) {
            v8h ov;
#pragma unroll
            for (int r = 0; r < 8; ++r)
                ov[r] = (_Float16)fmaxf(acc[t][r] * sA[r] + sB[r], 0.f);
            *(v8h*)(outb + (size_t)p * COUT) = ov;
        }
    }
}

// ---------------------------------------------------------------------------
// Stage 3: global mean pool NHWC (B, 900, 256) f16 -> (B, 256) f32
// ---------------------------------------------------------------------------
__global__ __launch_bounds__(256)
void meanpool_kernel(const _Float16* __restrict__ x, float* __restrict__ g, int HW) {
    int b = blockIdx.x, c = threadIdx.x;
    const _Float16* src = x + (size_t)b * HW * kFD + c;
    float s = 0.f;
    for (int p = 0; p < HW; ++p) s += (float)src[(size_t)p * kFD];
    g[(size_t)b * kFD + c] = s / (float)HW;
}

// ---------------------------------------------------------------------------
// Stage 4: lat = g @ mlp_w.T + mlp_b; LayerNorm(lat) -> d_out (B, 256) f32
// ---------------------------------------------------------------------------
__global__ __launch_bounds__(256)
void mlp_ln_kernel(const float* __restrict__ g,
                   const float* __restrict__ mw,   // (256, 256)
                   const float* __restrict__ mb,
                   const float* __restrict__ lg,
                   const float* __restrict__ lb,
                   float* __restrict__ out) {
    __shared__ float gg[kFD];
    __shared__ float red[kFD];
    int b = blockIdx.x, o = threadIdx.x;
    gg[o] = g[(size_t)b * kFD + o];
    __syncthreads();
    float acc = mb[o];
    const float* wrow = mw + (size_t)o * kFD;
#pragma unroll 4
    for (int k = 0; k < kFD; ++k) acc += wrow[k] * gg[k];

    red[o] = acc;
    __syncthreads();
    for (int off = 128; off > 0; off >>= 1) {
        if (o < off) red[o] += red[o + off];
        __syncthreads();
    }
    __shared__ float mu_s;
    if (o == 0) mu_s = red[0] / (float)kFD;
    __syncthreads();
    float d = acc - mu_s;
    red[o] = d * d;
    __syncthreads();
    for (int off = 128; off > 0; off >>= 1) {
        if (o < off) red[o] += red[o + off];
        __syncthreads();
    }
    __shared__ float var_s;
    if (o == 0) var_s = red[0] / (float)kFD;
    __syncthreads();
    out[(size_t)b * kFD + o] = d * rsqrtf(var_s + kEPS) * lg[o] + lb[o];
}

// ---------------------------------------------------------------------------
extern "C" void kernel_launch(void* const* d_in, const int* in_sizes, int n_in,
                              void* d_out, int out_size, void* d_ws, size_t ws_size,
                              hipStream_t stream) {
    (void)in_sizes; (void)n_in; (void)out_size; (void)ws_size;

    const float* pts    = (const float*)d_in[0];
    const float* pfn_w  = (const float*)d_in[1];
    const float* pfn_b  = (const float*)d_in[2];
    const float* pbn_g  = (const float*)d_in[3];
    const float* pbn_b  = (const float*)d_in[4];
    const float* pbn_m  = (const float*)d_in[5];
    const float* pbn_v  = (const float*)d_in[6];
    const float* cw[4]  = {(const float*)d_in[7],  (const float*)d_in[13],
                           (const float*)d_in[19], (const float*)d_in[25]};
    const float* cbias[4] = {(const float*)d_in[8],  (const float*)d_in[14],
                             (const float*)d_in[20], (const float*)d_in[26]};
    const float* bng[4] = {(const float*)d_in[9],  (const float*)d_in[15],
                           (const float*)d_in[21], (const float*)d_in[27]};
    const float* bnb[4] = {(const float*)d_in[10], (const float*)d_in[16],
                           (const float*)d_in[22], (const float*)d_in[28]};
    const float* bnm[4] = {(const float*)d_in[11], (const float*)d_in[17],
                           (const float*)d_in[23], (const float*)d_in[29]};
    const float* bnv[4] = {(const float*)d_in[12], (const float*)d_in[18],
                           (const float*)d_in[24], (const float*)d_in[30]};
    const float* mlp_w = (const float*)d_in[31];
    const float* mlp_b = (const float*)d_in[32];
    const float* ln_g  = (const float*)d_in[33];
    const float* ln_b  = (const float*)d_in[34];

    // Workspace bump allocator (256B-aligned slabs)
    char* ws = (char*)d_ws;
    size_t off = 0;
    auto alloc = [&](size_t bytes) -> void* {
        void* p = ws + off;
        off += (bytes + 255) & ~(size_t)255;
        return p;
    };
    float*    psum = (float*)alloc((size_t)kB * kP * 3 * 4);
    float*    pcnt = (float*)alloc((size_t)kB * kP * 4);
    float*    pmax = (float*)alloc((size_t)kB * kP * kD * 4);
    _Float16* w1   = (_Float16*)alloc((size_t)64  * 64  * 9 * 2);
    _Float16* w2   = (_Float16*)alloc((size_t)128 * 64  * 9 * 2);
    _Float16* w3   = (_Float16*)alloc((size_t)256 * 128 * 9 * 2);
    _Float16* w4   = (_Float16*)alloc((size_t)256 * 256 * 9 * 2);
    _Float16* bev  = (_Float16*)alloc((size_t)kB * 240 * 240 * 64  * 2);
    _Float16* o1   = (_Float16*)alloc((size_t)kB * 240 * 240 * 64  * 2);
    _Float16* o2   = (_Float16*)alloc((size_t)kB * 120 * 120 * 128 * 2);
    _Float16* o3   = (_Float16*)alloc((size_t)kB * 60  * 60  * 256 * 2);
    _Float16* o4   = (_Float16*)alloc((size_t)kB * 30  * 30  * 256 * 2);
    float*    gbuf = (float*)alloc((size_t)kB * kFD * 4);

    // Zero the accumulation buffers every call (graph-replay safe)
    hipMemsetAsync(psum, 0, (size_t)kB * kP * 3 * 4, stream);
    hipMemsetAsync(pcnt, 0, (size_t)kB * kP * 4, stream);
    hipMemsetAsync(pmax, 0, (size_t)kB * kP * kD * 4, stream);

    const int npts = kB * kN;
    scatter_mean_kernel<<<(npts + 255) / 256, 256, 0, stream>>>(pts, psum, pcnt);
    pfn_max_kernel<<<(npts + 255) / 256, 256, 0, stream>>>(
        pts, psum, pcnt, pfn_w, pfn_b, pbn_g, pbn_b, pbn_m, pbn_v, pmax);

    wperm_kernel<<<(64 * 64 * 9 + 255) / 256, 256, 0, stream>>>(cw[0], w1, 64, 64);
    wperm_kernel<<<(128 * 64 * 9 + 255) / 256, 256, 0, stream>>>(cw[1], w2, 64, 128);
    wperm_kernel<<<(256 * 128 * 9 + 255) / 256, 256, 0, stream>>>(cw[2], w3, 128, 256);
    wperm_kernel<<<(256 * 256 * 9 + 255) / 256, 256, 0, stream>>>(cw[3], w4, 256, 256);

    {
        size_t tot = (size_t)kB * kP * kD;
        bev_cast_kernel<<<(unsigned)((tot + 255) / 256), 256, 0, stream>>>(pmax, bev);
    }

    // conv1: 64 -> 64, 240x240, stride 1
    conv3x3_wmma<64, 64><<<dim3((57600 + 63) / 64, 64 / 16, kB), 32, 0, stream>>>(
        bev, w1, cbias[0], bng[0], bnb[0], bnm[0], bnv[0], o1, 240, 240, 240, 240, 1);
    // conv2: 64 -> 128, -> 120x120, stride 2
    conv3x3_wmma<64, 128><<<dim3((14400 + 63) / 64, 128 / 16, kB), 32, 0, stream>>>(
        o1, w2, cbias[1], bng[1], bnb[1], bnm[1], bnv[1], o2, 240, 240, 120, 120, 2);
    // conv3: 128 -> 256, -> 60x60, stride 2
    conv3x3_wmma<128, 256><<<dim3((3600 + 63) / 64, 256 / 16, kB), 32, 0, stream>>>(
        o2, w3, cbias[2], bng[2], bnb[2], bnm[2], bnv[2], o3, 120, 120, 60, 60, 2);
    // conv4: 256 -> 256, -> 30x30, stride 2
    conv3x3_wmma<256, 256><<<dim3((900 + 63) / 64, 256 / 16, kB), 32, 0, stream>>>(
        o3, w4, cbias[3], bng[3], bnb[3], bnm[3], bnv[3], o4, 60, 60, 30, 30, 2);

    meanpool_kernel<<<kB, 256, 0, stream>>>(o4, gbuf, 30 * 30);
    mlp_ln_kernel<<<kB, 256, 0, stream>>>(gbuf, mlp_w, mlp_b, ln_g, ln_b, (float*)d_out);
}
